// MultiHeadLatentAttention_63471026700718
// MI455X (gfx1250) — matrix operand order repaired and checked by reading, not compiled, not run on hardware
//
#include <hip/hip_runtime.h>
#include <hip/hip_bf16.h>

// MLA for MI455X (gfx1250, wave32). Matmuls: v_wmma_f32_16x16x32_bf16.
// GEMM B-tiles staged into LDS via the Tensor Data Mover (tensor_load_to_lds,
// TENSORcnt), double-buffered. B=2 S=2048 D=2048 H=16 HD=128 RD=64 QR=1536 KVR=512

typedef __attribute__((ext_vector_type(16))) __bf16       v16bf;
typedef __attribute__((ext_vector_type(8)))  float        v8f;
typedef __attribute__((ext_vector_type(4)))  unsigned int v4u;
typedef __attribute__((ext_vector_type(8)))  int          v8i;
typedef __attribute__((ext_vector_type(4)))  int          v4i;

#define WMMA_BF16(a, b, c) \
  __builtin_amdgcn_wmma_f32_16x16x32_bf16(false, (a), false, (b), (short)0, (c), false, false)

// ---- fragment loaders (wave32; contiguous k order, same for A and B) ------------------
// A 16x32 bf16: row=lane&15, elems j = K dim 16*(lane>>4)+j (16 contiguous).
__device__ __forceinline__ v16bf load_a_frag_f32(const float* tile, int ld, int k0, int lane) {
  const int l = lane & 15, hi = lane >> 4;
  const float* p = tile + (size_t)l * ld + k0 + hi * 16;
  v16bf a;
#pragma unroll
  for (int j = 0; j < 16; ++j) a[j] = (__bf16)p[j];
  return a;
}
__device__ __forceinline__ v16bf load_a_frag_bf16(const __bf16* tile, int ld, int k0, int lane) {
  const int l = lane & 15, hi = lane >> 4;
  const __bf16* p = tile + (size_t)l * ld + k0 + hi * 16;
  v16bf a;
#pragma unroll
  for (int j = 0; j < 16; ++j) a[j] = p[j];
  return a;
}
// B 32x16: lane&15 = output col n; elems j = K dim 16*(lane>>4)+j (16 contiguous).
__device__ __forceinline__ v16bf load_b_frag_f32(const float* rows, int ld, int k0, int lane) {
  const int n = lane & 15, hi = lane >> 4;
  const float* p = rows + (size_t)n * ld + k0 + hi * 16;
  v16bf b;
#pragma unroll
  for (int j = 0; j < 16; ++j) b[j] = (__bf16)p[j];
  return b;
}
__device__ __forceinline__ v16bf load_b_frag_bf16(const __bf16* rows, int ld, int k0, int lane) {
  const int n = lane & 15, hi = lane >> 4;
  const __bf16* p = rows + (size_t)n * ld + k0 + hi * 16;
  v16bf b;
#pragma unroll
  for (int j = 0; j < 16; ++j) b[j] = p[j];
  return b;
}

// ---- TDM: DMA a 64-row x 32-col f32 tile (row stride K elems) into LDS -------
// D# per CDNA5 ISA. LDS rows padded by TDM (pad_interval
// = 32 DWORDs, pad_amount = 2 DWORDs) -> LDS row stride 34 floats (bank spread).
#define BT_LDS_STRIDE 34
__device__ __forceinline__ void tdm_load_b_tile(const float* gsrc, unsigned int lds_byte_addr, int ldK) {
  unsigned long long ga = (unsigned long long)(const void*)gsrc;
  v4u g0;
  g0[0] = 1u;                                   // count=1, user descriptor
  g0[1] = lds_byte_addr;                        // LDS dest (bytes)
  g0[2] = (unsigned int)(ga & 0xffffffffull);   // global_addr[31:0]
  g0[3] = (unsigned int)((ga >> 32) & 0x01ffffffull) | (2u << 30);  // addr[56:32] | type=2
  const unsigned int td0 = 1u << 30;            // huge tensor dims: no OOB clamp
  const unsigned int td1 = 1u << 20;
  const unsigned int tile0 = 32, tile1 = 64;
  const unsigned long long str0 = (unsigned long long)ldK;   // dim0 stride (elems)
  v8i g1;
  g1[0] = (int)((2u << 16)                      // data_size = 4B
              | (1u << 20)                      // pad_enable
              | (4u << 22)                      // pad_interval code 4 -> 32 DWORDs
              | (1u << 25));                    // pad_amount  code 1 -> 2 DWORDs
  g1[1] = (int)((td0 & 0xffffu) << 16);                         // [63:48]=td0 lo
  g1[2] = (int)(((td0 >> 16) & 0xffffu) | ((td1 & 0xffffu) << 16));
  g1[3] = (int)(((td1 >> 16) & 0xffffu) | (tile0 << 16));       // [127:112]=tile_dim0
  g1[4] = (int)(tile1);                                         // tile_dim1, tile_dim2=0
  g1[5] = (int)(str0 & 0xffffffffull);
  g1[6] = (int)((str0 >> 32) & 0xffffull);                      // dim1 stride = 0
  g1[7] = 0;
  v4i gz4 = {0, 0, 0, 0};                       // 2D tensor: groups 2/3 unused
  v8i gz8 = {0, 0, 0, 0, 0, 0, 0, 0};
  __builtin_amdgcn_tensor_load_to_lds(g0, g1, gz4, gz4, gz8, 0);
}

// ---- generic GEMM: C[M,N] = A[M,K] * W[N,K]^T  (f32 in/out, bf16 WMMA) -------
// block = 8 waves, tile 128x64. B chunk (64x32 f32) DMA'd to LDS by TDM,
// double-buffered; A fragments loaded directly from global.
__global__ __launch_bounds__(256) void gemm_bf16_kernel(
    const float* __restrict__ A, const float* __restrict__ W, float* __restrict__ C,
    int M, int N, int K) {
  __shared__ float bt[2][64 * BT_LDS_STRIDE];   // only __shared__ object -> LDS base 0
  const int lane = threadIdx.x & 31, wv = threadIdx.x >> 5;
  const int m0 = blockIdx.x * 128 + wv * 16;
  const int n0 = blockIdx.y * 64;
  const unsigned int ldsb = (unsigned int)(64 * BT_LDS_STRIDE * 4);
  v8f acc0 = (v8f)0.f, acc1 = (v8f)0.f, acc2 = (v8f)0.f, acc3 = (v8f)0.f;
  const float* arow = A + (size_t)m0 * K;
  const float* wrow = W + (size_t)n0 * K;
  const int nk = K >> 5;

  if (wv == 0) tdm_load_b_tile(wrow, 0u, K);    // prologue: chunk 0 -> buf 0
  for (int kc = 0; kc < nk; ++kc) {
    const int k0 = kc << 5;
    if (wv == 0) {
      if (kc + 1 < nk) {
        tdm_load_b_tile(wrow + (size_t)(k0 + 32), ((kc + 1) & 1) ? ldsb : 0u, K);
        __builtin_amdgcn_s_wait_tensorcnt(1);   // chunk kc landed (in-order)
      } else {
        __builtin_amdgcn_s_wait_tensorcnt(0);
      }
    }
    __syncthreads();                            // buf[kc&1] visible to all waves
    const float* bbase = &bt[kc & 1][0];
    __builtin_prefetch(arow + (size_t)(lane & 15) * K + k0 + 64, 0, 3);
    v16bf af = load_a_frag_f32(arow, K, k0, lane);
    v16bf b0 = load_b_frag_f32(bbase +  0 * BT_LDS_STRIDE, BT_LDS_STRIDE, 0, lane);
    acc0 = WMMA_BF16(af, b0, acc0);
    v16bf b1 = load_b_frag_f32(bbase + 16 * BT_LDS_STRIDE, BT_LDS_STRIDE, 0, lane);
    acc1 = WMMA_BF16(af, b1, acc1);
    v16bf b2 = load_b_frag_f32(bbase + 32 * BT_LDS_STRIDE, BT_LDS_STRIDE, 0, lane);
    acc2 = WMMA_BF16(af, b2, acc2);
    v16bf b3 = load_b_frag_f32(bbase + 48 * BT_LDS_STRIDE, BT_LDS_STRIDE, 0, lane);
    acc3 = WMMA_BF16(af, b3, acc3);
    __syncthreads();                            // done reading buf[kc&1] before reuse
  }
  const int cn = lane & 15, hi = lane >> 4;
#pragma unroll
  for (int r = 0; r < 8; ++r) {
    size_t ro = (size_t)(m0 + r + 8 * hi) * N + n0 + cn;
    C[ro + 0] = acc0[r]; C[ro + 16] = acc1[r]; C[ro + 32] = acc2[r]; C[ro + 48] = acc3[r];
  }
}

// ---- in-place RMSNorm over rows (C = 1536 or 512) ----------------------------
__global__ __launch_bounds__(256) void rmsnorm_kernel(float* x, const float* __restrict__ w, int C) {
  __shared__ float red[256];
  const int tid = threadIdx.x;
  float* p = x + (size_t)blockIdx.x * C;
  const int n = C >> 8;           // <= 6
  float loc[6];
  float ss = 0.f;
  for (int i = 0; i < n; ++i) { loc[i] = p[tid + (i << 8)]; ss += loc[i] * loc[i]; }
  red[tid] = ss; __syncthreads();
  for (int o = 128; o > 0; o >>= 1) { if (tid < o) red[tid] += red[tid + o]; __syncthreads(); }
  const float rs = rsqrtf(red[0] / (float)C + 1e-6f);
  for (int i = 0; i < n; ++i) p[tid + (i << 8)] = loc[i] * rs * w[tid + (i << 8)];
}

// ---- pack kernels: bf16 Q/K (rope applied) and transposed V ------------------
// Qp/Kp layout: [(b*16+h)*2048 + s]*128 + d ; Vt layout: [(b*16+h)*128 + d]*2048 + s
__global__ __launch_bounds__(256) void pack_q_kernel(
    const float* __restrict__ qnope, const float* __restrict__ kroperaw,
    const float* __restrict__ cosT, const float* __restrict__ sinT, __bf16* __restrict__ Qp) {
  int idx = blockIdx.x * 256 + threadIdx.x;         // 2^23 total
  int d = idx & 127, s = (idx >> 7) & 2047, h = (idx >> 18) & 15, bb = idx >> 22;
  int row = bb * 2048 + s;
  float val;
  if (d < 64) {
    val = qnope[(size_t)row * 1024 + h * 64 + d];
  } else {                                          // Q rope half = rope(k_rope_raw), bcast over h
    int r = d - 64, i = r >> 1;
    float c = cosT[s * 32 + i], sn = sinT[s * 32 + i];
    float xr = kroperaw[(size_t)row * 64 + 2 * i];
    float xi = kroperaw[(size_t)row * 64 + 2 * i + 1];
    val = (r & 1) ? (xr * sn + xi * c) : (xr * c - xi * sn);
  }
  Qp[idx] = (__bf16)(val * 0.08838834764831845f);   // fold 1/sqrt(128)
}
__global__ __launch_bounds__(256) void pack_k_kernel(
    const float* __restrict__ knope, const float* __restrict__ qrope,
    const float* __restrict__ cosT, const float* __restrict__ sinT, __bf16* __restrict__ Kp) {
  int idx = blockIdx.x * 256 + threadIdx.x;
  int d = idx & 127, s = (idx >> 7) & 2047, h = (idx >> 18) & 15, bb = idx >> 22;
  int row = bb * 2048 + s;
  float val;
  if (d < 64) {
    val = knope[(size_t)row * 1024 + h * 64 + d];
  } else {                                          // K rope half = rope(q_rope) per head
    int r = d - 64, i = r >> 1;
    float c = cosT[s * 32 + i], sn = sinT[s * 32 + i];
    float xr = qrope[(size_t)row * 1024 + h * 64 + 2 * i];
    float xi = qrope[(size_t)row * 1024 + h * 64 + 2 * i + 1];
    val = (r & 1) ? (xr * sn + xi * c) : (xr * c - xi * sn);
  }
  Kp[idx] = (__bf16)val;
}
__global__ __launch_bounds__(256) void pack_vt_kernel(
    const float* __restrict__ vmat, __bf16* __restrict__ Vt) {
  int idx = blockIdx.x * 256 + threadIdx.x;
  int s = idx & 2047, d = (idx >> 11) & 127, h = (idx >> 18) & 15, bb = idx >> 22;
  Vt[idx] = (__bf16)vmat[((size_t)(bb * 2048 + s)) * 2048 + h * 128 + d];
}

// ---- flash attention: 1 wave = 16 queries, keys streamed in chunks of 32 -----
__global__ __launch_bounds__(128) void attn_kernel(
    const __bf16* __restrict__ Qp, const __bf16* __restrict__ Kp,
    const __bf16* __restrict__ Vt, float* __restrict__ out) {
  __shared__ float pbuf[4][16 * 32];                // per-wave P staging
  const int lane = threadIdx.x & 31, wv = threadIdx.x >> 5;
  const int h = blockIdx.y, b = blockIdx.z;
  const int q0 = (blockIdx.x * 4 + wv) * 16;
  const int bh = b * 16 + h;
  const __bf16* qbase = Qp + ((size_t)bh * 2048 + q0) * 128;
  const __bf16* kbase = Kp + (size_t)bh * 2048 * 128;
  const __bf16* vbase = Vt + (size_t)bh * 128 * 2048;
  const int hi = lane >> 4, cn = lane & 15;

  v16bf aq[4];
#pragma unroll
  for (int c = 0; c < 4; ++c) aq[c] = load_a_frag_bf16(qbase, 128, c * 32, lane);

  v8f o[8];
#pragma unroll
  for (int i = 0; i < 8; ++i) o[i] = (v8f)0.f;
  float mrow[8], lrow[8];
#pragma unroll
  for (int r = 0; r < 8; ++r) { mrow[r] = -3e38f; lrow[r] = 0.f; }

  float* myp = &pbuf[wv][0];
  const int nchunk = ((q0 + 15) >> 5) + 1;
  for (int kc = 0; kc < nchunk; ++kc) {
    const int k0 = kc * 32;
    v8f s0 = (v8f)0.f, s1 = (v8f)0.f;               // keys k0..+15 / k0+16..+31
#pragma unroll
    for (int c = 0; c < 4; ++c) {
      v16bf bk0 = load_b_frag_bf16(kbase + (size_t)k0 * 128, 128, c * 32, lane);
      s0 = WMMA_BF16(aq[c], bk0, s0);
      v16bf bk1 = load_b_frag_bf16(kbase + (size_t)(k0 + 16) * 128, 128, c * 32, lane);
      s1 = WMMA_BF16(aq[c], bk1, s1);
    }
    if (k0 + 31 > q0) {                             // causal mask on diagonal chunks
#pragma unroll
      for (int r = 0; r < 8; ++r) {
        const int qrow = q0 + r + 8 * hi;
        if (k0 + cn > qrow)      s0[r] = -3e38f;
        if (k0 + 16 + cn > qrow) s1[r] = -3e38f;
      }
    }
#pragma unroll
    for (int r = 0; r < 8; ++r) {                   // online softmax, row = q0+r+8*hi
      float v = fmaxf(s0[r], s1[r]);
#pragma unroll
      for (int off = 1; off < 16; off <<= 1) v = fmaxf(v, __shfl_xor(v, off, 32));
      const float newm = fmaxf(mrow[r], v);
      const float corr = __expf(mrow[r] - newm);
      const float p0 = __expf(s0[r] - newm);
      const float p1 = __expf(s1[r] - newm);
      float rsum = p0 + p1;
#pragma unroll
      for (int off = 1; off < 16; off <<= 1) rsum += __shfl_xor(rsum, off, 32);
      lrow[r] = lrow[r] * corr + rsum;
      mrow[r] = newm;
#pragma unroll
      for (int dt = 0; dt < 8; ++dt) o[dt][r] *= corr;
      const int prow = r + 8 * hi;
      myp[prow * 32 + cn] = p0;
      myp[prow * 32 + 16 + cn] = p1;
    }
    asm volatile("s_wait_dscnt 0" ::: "memory");    // DS in-order per wave
    v16bf pa = load_a_frag_f32(myp, 32, 0, lane);   // P as 16x32 A-fragment (bf16)
#pragma unroll
    for (int dt = 0; dt < 8; ++dt) {
      v16bf bv = load_b_frag_bf16(vbase + (size_t)(dt * 16) * 2048, 2048, k0, lane);
      o[dt] = WMMA_BF16(pa, bv, o[dt]);
    }
  }
#pragma unroll
  for (int dt = 0; dt < 8; ++dt)
#pragma unroll
    for (int r = 0; r < 8; ++r) {
      const int s = q0 + r + 8 * hi;
      out[((size_t)b * 2048 + s) * 2048 + h * 128 + dt * 16 + cn] = o[dt][r] / lrow[r];
    }
}

// ---- host side ----------------------------------------------------------------
extern "C" void kernel_launch(void* const* d_in, const int* in_sizes, int n_in,
                              void* d_out, int out_size, void* d_ws, size_t ws_size,
                              hipStream_t stream) {
  const float* x         = (const float*)d_in[0];
  const float* fcos      = (const float*)d_in[1];
  const float* fsin      = (const float*)d_in[2];
  const float* w_cq      = (const float*)d_in[3];
  const float* w_dq_n    = (const float*)d_in[4];
  const float* w_dq_r    = (const float*)d_in[5];
  const float* w_ckv     = (const float*)d_in[6];
  const float* w_dk_n    = (const float*)d_in[7];
  const float* w_dv      = (const float*)d_in[8];
  const float* w_krope   = (const float*)d_in[9];
  const float* w_proj    = (const float*)d_in[10];
  const float* q_norm_w  = (const float*)d_in[11];
  const float* kv_norm_w = (const float*)d_in[12];

  const int R = 4096;                                // B*S rows
  float* cq    = (float*)d_ws;                       // R*1536 (normed in place)
  float* ckv   = cq    + (size_t)R * 1536;           // R*512  (normed in place)
  float* krr   = ckv   + (size_t)R * 512;            // R*64
  float* qnope = krr   + (size_t)R * 64;             // R*1024
  float* qrope = qnope + (size_t)R * 1024;           // R*1024
  float* knope = qrope + (size_t)R * 1024;           // R*1024
  float* vmat  = knope + (size_t)R * 1024;           // R*2048
  float* attn  = vmat  + (size_t)R * 2048;           // R*2048
  __bf16* Qp = (__bf16*)(attn + (size_t)R * 2048);   // 2^23 bf16 each
  __bf16* Kp = Qp + (size_t)(1u << 23);
  __bf16* Vt = Kp + (size_t)(1u << 23);

  const dim3 blk(256);
  // down projections
  gemm_bf16_kernel<<<dim3(32, 24), blk, 0, stream>>>(x, w_cq,    cq,  R, 1536, 2048);
  gemm_bf16_kernel<<<dim3(32,  8), blk, 0, stream>>>(x, w_ckv,   ckv, R,  512, 2048);
  gemm_bf16_kernel<<<dim3(32,  1), blk, 0, stream>>>(x, w_krope, krr, R,   64, 2048);
  // rmsnorm (in place)
  rmsnorm_kernel<<<R, 256, 0, stream>>>(cq,  q_norm_w,  1536);
  rmsnorm_kernel<<<R, 256, 0, stream>>>(ckv, kv_norm_w,  512);
  // up projections
  gemm_bf16_kernel<<<dim3(32, 16), blk, 0, stream>>>(cq,  w_dq_n, qnope, R, 1024, 1536);
  gemm_bf16_kernel<<<dim3(32, 16), blk, 0, stream>>>(cq,  w_dq_r, qrope, R, 1024, 1536);
  gemm_bf16_kernel<<<dim3(32, 16), blk, 0, stream>>>(ckv, w_dk_n, knope, R, 1024,  512);
  gemm_bf16_kernel<<<dim3(32, 32), blk, 0, stream>>>(ckv, w_dv,   vmat,  R, 2048,  512);
  // rope + bf16 packing
  pack_q_kernel <<<32768, 256, 0, stream>>>(qnope, krr, fcos, fsin, Qp);
  pack_k_kernel <<<32768, 256, 0, stream>>>(knope, qrope, fcos, fsin, Kp);
  pack_vt_kernel<<<32768, 256, 0, stream>>>(vmat, Vt);
  // flash attention
  attn_kernel<<<dim3(32, 16, 2), 128, 0, stream>>>(Qp, Kp, Vt, attn);
  // output projection -> d_out
  gemm_bf16_kernel<<<dim3(32, 32), blk, 0, stream>>>(attn, w_proj, (float*)d_out, R, 2048, 2048);
}
